// GPT_36309653520994
// MI455X (gfx1250) — matrix operand order repaired and checked by reading, not compiled, not run on hardware
//
#include <hip/hip_runtime.h>
#include <hip/hip_bf16.h>
#include <math.h>

// ---------------- model constants ----------------
enum : int { NB = 4, NT = 1024, NC = 768, NH = 12, ND = 64, NLAY = 6 };
enum : int { NV = 50257, NVP = 50304, NM = NB * NT };   // NVP = NV padded to 128

typedef __attribute__((ext_vector_type(16))) _Float16 v16h;
typedef __attribute__((ext_vector_type(8)))  _Float16 v8h;
typedef __attribute__((ext_vector_type(8)))  float    v8f;

// =====================================================================
// Embedding: x[b,t,:] = tok_emb[idx[b,t],:] + pos_emb[t,:]   (fp32)
// =====================================================================
__global__ __launch_bounds__(256)
void embed_kernel(const int* __restrict__ idx, const float* __restrict__ tok,
                  const float* __restrict__ pos, float* __restrict__ x)
{
    const int row = blockIdx.x;           // 0..NM-1
    const int t   = row & (NT - 1);
    const int token = idx[row];
    const float* tp = tok + (size_t)token * NC;
    const float* pp = pos + (size_t)t * NC;
    float* xp = x + (size_t)row * NC;
    #pragma unroll
    for (int j = 0; j < 3; ++j) {
        int c = threadIdx.x + j * 256;
        xp[c] = tp[c] + pp[c];
    }
}

// =====================================================================
// Transpose + convert: Wt[n,k] (f16, n padded with zeros) = W[k,n] (f32)
// LDS-tiled so both global access directions are coalesced.
// Grid: (K/32, Npad/32), 256 threads.
// =====================================================================
__global__ __launch_bounds__(256)
void transpose_cvt_kernel(const float* __restrict__ W, _Float16* __restrict__ Wt,
                          int K, int N)
{
    __shared__ float tile[32][33];
    const int kt = blockIdx.x * 32;
    const int nt = blockIdx.y * 32;
    const int tx = threadIdx.x & 31;
    const int ty = threadIdx.x >> 5;      // 0..7
    const int n  = nt + tx;
    #pragma unroll
    for (int j = 0; j < 4; ++j) {
        const int k = kt + ty + j * 8;
        tile[ty + j * 8][tx] = (n < N) ? W[(size_t)k * N + n] : 0.0f;
    }
    __syncthreads();
    #pragma unroll
    for (int j = 0; j < 4; ++j) {
        const int nrow = nt + ty + j * 8;
        Wt[(size_t)nrow * K + kt + tx] = (_Float16)tile[tx][ty + j * 8];
    }
}

// =====================================================================
// LayerNorm (unbiased std, divide by std+eps, per reference). f16 out.
// =====================================================================
__global__ __launch_bounds__(256)
void ln_kernel(const float* __restrict__ x, const float* __restrict__ g,
               const float* __restrict__ be, _Float16* __restrict__ y)
{
    __shared__ float red[256];
    const int row = blockIdx.x;
    const int tid = threadIdx.x;
    const float* xr = x + (size_t)row * NC;

    float v[3];
    float s = 0.f;
    #pragma unroll
    for (int j = 0; j < 3; ++j) { v[j] = xr[tid + j * 256]; s += v[j]; }
    red[tid] = s; __syncthreads();
    for (int st = 128; st > 0; st >>= 1) {
        if (tid < st) red[tid] += red[tid + st];
        __syncthreads();
    }
    const float mean = red[0] * (1.0f / NC);
    __syncthreads();

    float sq = 0.f;
    #pragma unroll
    for (int j = 0; j < 3; ++j) { float d = v[j] - mean; sq += d * d; }
    red[tid] = sq; __syncthreads();
    for (int st = 128; st > 0; st >>= 1) {
        if (tid < st) red[tid] += red[tid + st];
        __syncthreads();
    }
    const float var = red[0] * (1.0f / (NC - 1));      // ddof=1
    const float inv = 1.0f / (sqrtf(var) + 1e-6f);     // g*(x-mean)/(std+eps)+b

    _Float16* yr = y + (size_t)row * NC;
    #pragma unroll
    for (int j = 0; j < 3; ++j) {
        int c = tid + j * 256;
        yr[c] = (_Float16)(g[c] * (v[j] - mean) * inv + be[c]);
    }
}

// =====================================================================
// GEMM: out[M,N] = A[M,K] * Bt[N,K]^T (+bias) (+residual) (relu?)
// A, Bt f16 in global. Double-buffered software pipeline:
//   tile i+1 is copied global->LDS with GLOBAL_LOAD_ASYNC_TO_LDS_B128
//   (ASYNCcnt) while the 8 v_wmma_f32_16x16x32_f16 of tile i execute.
// One barrier per k-step; s_wait_asynccnt 0 covers exactly the
// previous issue (in-order completion). Block tile 128x128, 8 waves
// as 4x2, wave tile 32x64, K-step 32. Bt rows pre-padded => no guards.
// =====================================================================
__global__ __launch_bounds__(256)
void gemm_kernel(const _Float16* __restrict__ A, const _Float16* __restrict__ Bt,
                 const float* __restrict__ bias, const float* __restrict__ res,
                 float* __restrict__ out32, _Float16* __restrict__ out16,
                 int M, int N, int K, int relu)
{
    constexpr int BM = 128, BN = 128, BK = 32;
    __shared__ __align__(64) _Float16 As[2][BM * BK];   // 2 x 8 KB
    __shared__ __align__(64) _Float16 Bst[2][BN * BK];  // 2 x 8 KB

    const int tid  = threadIdx.x;
    const int lane = tid & 31;
    const int wave = tid >> 5;         // 0..7
    const int wm   = wave >> 1;        // 0..3 (wave row)
    const int wn   = wave & 1;         // 0..1 (wave col, 64 wide)
    const int nl   = lane & 15;
    const int hi   = lane >> 4;

    const int m0 = blockIdx.y * BM;
    const int n0 = blockIdx.x * BN;
    const bool full = (n0 + BN) <= N;  // uniform: only head edge block is false

    v8f acc[2][4] = {};

    // async staging: each thread copies 2x16B for A and 2x16B for B per tile
    const int r0 = tid >> 2;            // 0..63
    const int sg = (tid & 3) * 8;       // 0,8,16,24
    unsigned la0[2], la1[2], lb0[2], lb1[2];
    #pragma unroll
    for (int b = 0; b < 2; ++b) {
        la0[b] = (unsigned)(size_t)&As[b][r0 * BK + sg];
        la1[b] = (unsigned)(size_t)&As[b][(r0 + 64) * BK + sg];
        lb0[b] = (unsigned)(size_t)&Bst[b][r0 * BK + sg];
        lb1[b] = (unsigned)(size_t)&Bst[b][(r0 + 64) * BK + sg];
    }
    // incrementing global pointers (advance BK per issued tile)
    const _Float16* pa0 = A  + (size_t)(m0 + r0)      * K + sg;
    const _Float16* pa1 = A  + (size_t)(m0 + r0 + 64) * K + sg;
    const _Float16* pb0 = Bt + (size_t)(n0 + r0)      * K + sg;
    const _Float16* pb1 = Bt + (size_t)(n0 + r0 + 64) * K + sg;

    auto issue_tile = [&](int buf) {
        asm volatile("global_load_async_to_lds_b128 %0, %1, off"
                     :: "v"(la0[buf]), "v"(pa0) : "memory");
        asm volatile("global_load_async_to_lds_b128 %0, %1, off"
                     :: "v"(la1[buf]), "v"(pa1) : "memory");
        asm volatile("global_load_async_to_lds_b128 %0, %1, off"
                     :: "v"(lb0[buf]), "v"(pb0) : "memory");
        asm volatile("global_load_async_to_lds_b128 %0, %1, off"
                     :: "v"(lb1[buf]), "v"(pb1) : "memory");
        pa0 += BK; pa1 += BK; pb0 += BK; pb1 += BK;
    };

    issue_tile(0);                              // prologue: tile 0 -> buf 0
    const int niter = K / BK;
    for (int it = 0; it < niter; ++it) {
        const int cur = it & 1;
        // own copies of tile `it` complete (only ones outstanding)
        asm volatile("s_wait_asynccnt 0x0" ::: "memory");
        // all waves: tile `it` visible; everyone done reading tile `it-1`
        __syncthreads();
        if (it + 1 < niter) {
            issue_tile(cur ^ 1);                // overlap copy of tile it+1
            __builtin_prefetch(pa0, 0, 1);      // CDNA5 global_prefetch (it+2)
            __builtin_prefetch(pb0, 0, 1);
        }

        // ---- A fragments: lane nl = row m; k = hi*16 + e (contiguous) ----
        v16h af[2];
        #pragma unroll
        for (int i = 0; i < 2; ++i) {
            const int m = wm * 32 + i * 16 + nl;
            v8h lo = *(const v8h*)&As[cur][m * BK + hi * 16];
            v8h hh = *(const v8h*)&As[cur][m * BK + hi * 16 + 8];
            #pragma unroll
            for (int e = 0; e < 8; ++e) { af[i][e] = lo[e]; af[i][8 + e] = hh[e]; }
        }
        // ---- B fragments: lane nl = col n; k = hi*16 + i (contiguous) ----
        #pragma unroll
        for (int jn = 0; jn < 4; ++jn) {
            const int n = wn * 64 + jn * 16 + nl;
            v16h bf = *(const v16h*)&Bst[cur][n * BK + hi * 16];
            #pragma unroll
            for (int i = 0; i < 2; ++i)
                acc[i][jn] = __builtin_amdgcn_wmma_f32_16x16x32_f16(
                    false, af[i], false, bf, (short)0, acc[i][jn], false, false);
        }
    }

    // ---- epilogue: C/D layout lane nl = col, m = hi*8 + r ----
    #pragma unroll
    for (int i = 0; i < 2; ++i) {
        #pragma unroll
        for (int jn = 0; jn < 4; ++jn) {
            const int n = n0 + wn * 64 + jn * 16 + nl;
            if (full || n < N) {
                #pragma unroll
                for (int r = 0; r < 8; ++r) {
                    const int m = m0 + wm * 32 + i * 16 + hi * 8 + r;
                    float v = acc[i][jn][r];
                    if (bias) v += bias[n];
                    if (res)  v += res[(size_t)m * N + n];
                    if (relu) v = fmaxf(v, 0.0f);
                    if (out16) out16[(size_t)m * N + n] = (_Float16)v;
                    else       out32[(size_t)m * N + n] = v;
                }
            }
        }
    }
}

// =====================================================================
// Split fused QKV (f16 [M,3C]) into q[B,H,T,D] (pre-scaled by 1/8),
// k[B,H,T,D], and vT[B,H,D,T] (transposed for contiguous PV B-fragments).
// =====================================================================
__global__ __launch_bounds__(256)
void qkvsplit_kernel(const _Float16* __restrict__ qkv, _Float16* __restrict__ qh,
                     _Float16* __restrict__ kh, _Float16* __restrict__ vT)
{
    const int row = blockIdx.x;          // b*T + t
    const int b = row >> 10;             // /NT
    const int t = row & (NT - 1);
    const _Float16* rp = qkv + (size_t)row * (3 * NC);
    for (int c = threadIdx.x; c < 3 * NC; c += 256) {
        const int which = c / NC;        // 0=q 1=k 2=v
        const int cc = c - which * NC;
        const int h = cc >> 6;           // /ND
        const int d = cc & (ND - 1);
        const int bh = b * NH + h;
        const _Float16 v = rp[c];
        if (which == 0)      qh[((size_t)bh * NT + t) * ND + d] = (_Float16)((float)v * 0.125f);
        else if (which == 1) kh[((size_t)bh * NT + t) * ND + d] = v;
        else                 vT[((size_t)bh * ND + d) * NT + t] = v;
    }
}

// =====================================================================
// Causal flash attention. Grid: (B*H, T/128). 8 waves/block, each wave
// owns 16 query rows. QK^T and P*V via WMMA; online softmax via
// shfl_xor butterflies (masks 1..8 keep the lane-half / C-layout rows).
// Context written directly in f16 for the projection GEMM.
// =====================================================================
__global__ __launch_bounds__(256)
void attn_kernel(const _Float16* __restrict__ qh, const _Float16* __restrict__ kh,
                 const _Float16* __restrict__ vT, _Float16* __restrict__ ctx)
{
    __shared__ __align__(64) _Float16 Pst[8][16 * 32];  // per-wave P staging
    const int bh = blockIdx.x;
    const int b = bh / NH, h = bh % NH;
    const int wave = threadIdx.x >> 5;
    const int lane = threadIdx.x & 31;
    const int nl = lane & 15, hi = lane >> 4;
    const int q0 = blockIdx.y * 128 + wave * 16;

    const _Float16* Q  = qh + (size_t)bh * NT * ND;
    const _Float16* Kk = kh + (size_t)bh * NT * ND;
    const _Float16* Vt = vT + (size_t)bh * ND * NT;

    // Q A-fragments (K-dim = head dim, two halves of D=64)
    v16h aq[2];
    #pragma unroll
    for (int c = 0; c < 2; ++c) {
        const _Float16* qp = Q + (size_t)(q0 + nl) * ND + c * 32;
        v8h lo = *(const v8h*)(qp + hi * 16);
        v8h hh = *(const v8h*)(qp + hi * 16 + 8);
        #pragma unroll
        for (int e = 0; e < 8; ++e) { aq[c][e] = lo[e]; aq[c][8 + e] = hh[e]; }
    }

    v8f o[4] = {};                 // 16x64 output: 4 d-tiles of 16
    float mrun[8], lrun[8];
    #pragma unroll
    for (int r = 0; r < 8; ++r) { mrun[r] = -1e30f; lrun[r] = 0.f; }

    const float NEGINF = -__builtin_inff();
    const int nkt = (q0 + 16 + 31) >> 5;       // #32-key tiles (causal bound)
    for (int jt = 0; jt < nkt; ++jt) {
        const int kb = jt * 32;
        // ---- S = Q K^T (scale already folded into q): two 16-key subtiles ----
        v8f s[2] = {};
        #pragma unroll
        for (int c16 = 0; c16 < 2; ++c16) {
            const int keyn = kb + c16 * 16 + nl;
            #pragma unroll
            for (int c = 0; c < 2; ++c) {
                v16h bk = *(const v16h*)(Kk + (size_t)keyn * ND + c * 32 + hi * 16);
                s[c16] = __builtin_amdgcn_wmma_f32_16x16x32_f16(
                    false, aq[c], false, bk, (short)0, s[c16], false, false);
            }
        }
        // ---- online softmax (rows live in hi-half; butterflies stay in half) ----
        #pragma unroll
        for (int r = 0; r < 8; ++r) {
            const int qrow = q0 + hi * 8 + r;
            float s0 = (kb + nl      <= qrow) ? s[0][r] : NEGINF;
            float s1 = (kb + 16 + nl <= qrow) ? s[1][r] : NEGINF;
            float mt = fmaxf(s0, s1);
            #pragma unroll
            for (int mk = 1; mk < 16; mk <<= 1) mt = fmaxf(mt, __shfl_xor(mt, mk, 32));
            const float mnew = fmaxf(mrun[r], mt);
            const float p0 = __expf(s0 - mnew);
            const float p1 = __expf(s1 - mnew);
            float lt = p0 + p1;
            #pragma unroll
            for (int mk = 1; mk < 16; mk <<= 1) lt += __shfl_xor(lt, mk, 32);
            const float scale = __expf(mrun[r] - mnew);
            lrun[r] = lrun[r] * scale + lt;
            mrun[r] = mnew;
            #pragma unroll
            for (int t = 0; t < 4; ++t) o[t][r] *= scale;
            Pst[wave][(hi * 8 + r) * 32 + nl]      = (_Float16)p0;
            Pst[wave][(hi * 8 + r) * 32 + 16 + nl] = (_Float16)p1;
        }
        // ---- reload P in A-fragment layout (same-wave LDS, in-order) ----
        v16h ap;
        {
            const _Float16* pp = &Pst[wave][nl * 32];
            v8h lo = *(const v8h*)(pp + hi * 16);
            v8h hh = *(const v8h*)(pp + hi * 16 + 8);
            #pragma unroll
            for (int e = 0; e < 8; ++e) { ap[e] = lo[e]; ap[8 + e] = hh[e]; }
        }
        // ---- O += P * V  (vT gives contiguous 32B B-fragments) ----
        #pragma unroll
        for (int t = 0; t < 4; ++t) {
            v16h bv = *(const v16h*)(Vt + (size_t)(t * 16 + nl) * NT + kb + hi * 16);
            o[t] = __builtin_amdgcn_wmma_f32_16x16x32_f16(
                false, ap, false, bv, (short)0, o[t], false, false);
        }
    }

    // ---- write context in [B,T,H*D] layout (f16) for the projection GEMM ----
    #pragma unroll
    for (int t = 0; t < 4; ++t) {
        #pragma unroll
        for (int r = 0; r < 8; ++r) {
            const int trow = q0 + hi * 8 + r;
            ctx[((size_t)(b * NT + trow)) * NC + h * ND + t * 16 + nl] =
                (_Float16)(o[t][r] / lrun[r]);
        }
    }
}

// =====================================================================
// Per-row cross-entropy: rowloss[m] = logsumexp(logits[m,:]) - logits[m,tgt]
// =====================================================================
__global__ __launch_bounds__(256)
void rowloss_kernel(const float* __restrict__ logits, const int* __restrict__ tgt,
                    float* __restrict__ rowloss)
{
    __shared__ float rm[256], rs[256];
    const int row = blockIdx.x;
    const int tid = threadIdx.x;
    const float* lr = logits + (size_t)row * NV;

    float m = -__builtin_inff(), s = 0.f;
    for (int c = tid; c < NV; c += 256) {
        const float v = lr[c];
        if (v > m) { s = s * __expf(m - v) + 1.0f; m = v; }
        else       { s += __expf(v - m); }
    }
    rm[tid] = m; rs[tid] = s; __syncthreads();
    for (int st = 128; st > 0; st >>= 1) {
        if (tid < st) {
            const float m2 = rm[tid + st], s2 = rs[tid + st];
            const float Mx = fmaxf(rm[tid], m2);
            rs[tid] = rs[tid] * __expf(rm[tid] - Mx) + s2 * __expf(m2 - Mx);
            rm[tid] = Mx;
        }
        __syncthreads();
    }
    if (tid == 0) {
        const float lse = rm[0] + __logf(rs[0]);
        rowloss[row] = lse - lr[tgt[row]];
    }
}

__global__ __launch_bounds__(256)
void meanloss_kernel(const float* __restrict__ rowloss, float* __restrict__ loss)
{
    __shared__ float red[256];
    const int tid = threadIdx.x;
    float s = 0.f;
    for (int i = tid; i < NM; i += 256) s += rowloss[i];
    red[tid] = s; __syncthreads();
    for (int st = 128; st > 0; st >>= 1) {
        if (tid < st) red[tid] += red[tid + st];
        __syncthreads();
    }
    if (tid == 0) loss[0] = red[0] * (1.0f / NM);
}

// =====================================================================
// Host: full forward pass, all on `stream`, scratch carved from d_ws.
// =====================================================================
extern "C" void kernel_launch(void* const* d_in, const int* in_sizes, int n_in,
                              void* d_out, int out_size, void* d_ws, size_t ws_size,
                              hipStream_t stream)
{
    (void)in_sizes; (void)n_in; (void)out_size; (void)ws_size;
    const int*   idx  = (const int*)d_in[0];
    const int*   tgt  = (const int*)d_in[1];
    const float* tok  = (const float*)d_in[2];
    const float* pos  = (const float*)d_in[3];
    const float* Wqkv = (const float*)d_in[4];
    const float* Wo   = (const float*)d_in[5];
    const float* bo   = (const float*)d_in[6];
    const float* W1   = (const float*)d_in[7];
    const float* b1   = (const float*)d_in[8];
    const float* W2   = (const float*)d_in[9];
    const float* b2   = (const float*)d_in[10];
    const float* g1   = (const float*)d_in[11];
    const float* be1  = (const float*)d_in[12];
    const float* g2   = (const float*)d_in[13];
    const float* be2  = (const float*)d_in[14];
    const float* gf   = (const float*)d_in[15];
    const float* bef  = (const float*)d_in[16];
    const float* Wh   = (const float*)d_in[17];
    const float* bhd  = (const float*)d_in[18];

    float* logits = (float*)d_out;
    float* loss   = logits + (size_t)NM * NV;

    char* p = (char*)d_ws;
    auto carve = [&](size_t bytes) -> char* {
        char* r = p; p += (bytes + 255) & ~(size_t)255; return r;
    };
    // f16 transposed weights (per layer contiguous)
    _Float16* WqkvT = (_Float16*)carve((size_t)NLAY * 3 * NC * NC * 2);
    _Float16* WoT   = (_Float16*)carve((size_t)NLAY * NC * NC * 2);
    _Float16* W1T   = (_Float16*)carve((size_t)NLAY * 4 * NC * NC * 2);
    _Float16* W2T   = (_Float16*)carve((size_t)NLAY * 4 * NC * NC * 2);
    _Float16* WhT   = (_Float16*)carve((size_t)NVP * NC * 2);
    // activations
    float*    x    = (float*)carve((size_t)NM * NC * 4);
    _Float16* xln  = (_Float16*)carve((size_t)NM * NC * 2);
    _Float16* qkvh = (_Float16*)carve((size_t)NM * 3 * NC * 2);
    _Float16* qh   = (_Float16*)carve((size_t)NM * NC * 2);
    _Float16* kh   = (_Float16*)carve((size_t)NM * NC * 2);
    _Float16* vT   = (_Float16*)carve((size_t)NM * NC * 2);
    _Float16* ctxh = (_Float16*)carve((size_t)NM * NC * 2);
    _Float16* h1h  = (_Float16*)carve((size_t)NM * 4 * NC * 2);
    float*    rls  = (float*)carve((size_t)NM * 4);

    const dim3 blk(256);

    // ---- one-time (per call) weight transpose + f16 conversion ----
    for (int l = 0; l < NLAY; ++l) {
        transpose_cvt_kernel<<<dim3(NC / 32, 3 * NC / 32), blk, 0, stream>>>(
            Wqkv + (size_t)l * NC * 3 * NC, WqkvT + (size_t)l * 3 * NC * NC, NC, 3 * NC);
        transpose_cvt_kernel<<<dim3(NC / 32, NC / 32), blk, 0, stream>>>(
            Wo + (size_t)l * NC * NC, WoT + (size_t)l * NC * NC, NC, NC);
        transpose_cvt_kernel<<<dim3(NC / 32, 4 * NC / 32), blk, 0, stream>>>(
            W1 + (size_t)l * NC * 4 * NC, W1T + (size_t)l * 4 * NC * NC, NC, 4 * NC);
        transpose_cvt_kernel<<<dim3(4 * NC / 32, NC / 32), blk, 0, stream>>>(
            W2 + (size_t)l * 4 * NC * NC, W2T + (size_t)l * 4 * NC * NC, 4 * NC, NC);
    }
    transpose_cvt_kernel<<<dim3(NC / 32, NVP / 32), blk, 0, stream>>>(Wh, WhT, NC, NV);

    embed_kernel<<<NM, blk, 0, stream>>>(idx, tok, pos, x);

    for (int l = 0; l < NLAY; ++l) {
        ln_kernel<<<NM, blk, 0, stream>>>(x, g1 + l * NC, be1 + l * NC, xln);
        gemm_kernel<<<dim3(3 * NC / 128, NM / 128), blk, 0, stream>>>(
            xln, WqkvT + (size_t)l * 3 * NC * NC, nullptr, nullptr, nullptr, qkvh,
            NM, 3 * NC, NC, 0);
        qkvsplit_kernel<<<NM, blk, 0, stream>>>(qkvh, qh, kh, vT);
        attn_kernel<<<dim3(NB * NH, NT / 128), blk, 0, stream>>>(qh, kh, vT, ctxh);
        gemm_kernel<<<dim3(NC / 128, NM / 128), blk, 0, stream>>>(
            ctxh, WoT + (size_t)l * NC * NC, bo + l * NC, x, x, nullptr,
            NM, NC, NC, 0);
        ln_kernel<<<NM, blk, 0, stream>>>(x, g2 + l * NC, be2 + l * NC, xln);
        gemm_kernel<<<dim3(4 * NC / 128, NM / 128), blk, 0, stream>>>(
            xln, W1T + (size_t)l * 4 * NC * NC, b1 + (size_t)l * 4 * NC, nullptr,
            nullptr, h1h, NM, 4 * NC, NC, 1);
        gemm_kernel<<<dim3(NC / 128, NM / 128), blk, 0, stream>>>(
            h1h, W2T + (size_t)l * 4 * NC * NC, b2 + l * NC, x, x, nullptr,
            NM, NC, 4 * NC, 0);
    }

    ln_kernel<<<NM, blk, 0, stream>>>(x, gf, bef, xln);
    gemm_kernel<<<dim3(NVP / 128, NM / 128), blk, 0, stream>>>(
        xln, WhT, bhd, nullptr, logits, nullptr, NM, NV, NC, 0);
    rowloss_kernel<<<NM, blk, 0, stream>>>(logits, tgt, rls);
    meanloss_kernel<<<1, blk, 0, stream>>>(rls, loss);
}
